// MashDecoder_28424093565554
// MI455X (gfx1250) — matrix operand order repaired and checked by reading, not compiled, not run on hardware
//
#include <hip/hip_runtime.h>
#include <hip/hip_bf16.h>

// ---------------------------------------------------------------------------
// MashDecoder for MI455X (gfx1250): WMMA bf16 GEMMs + wave32 VALU kernels.
// ---------------------------------------------------------------------------

typedef __attribute__((ext_vector_type(8)))  float  v8f;
typedef __attribute__((ext_vector_type(16))) __bf16 v16bf;
typedef __attribute__((ext_vector_type(2)))  __bf16 bf16x2;
typedef __attribute__((ext_vector_type(4)))  float  vf4;

#define D_MODEL 400
#define D_INNER 800
#define D_STATE 16
#define DT_RANK 25
#define N_LAYER 24
#define HEADS 4
#define DIM_HEAD 400
#define Bb 4
#define N_ANCHOR 400
#define N_QRY 4096
#define BL (Bb * N_ANCHOR)        /* 1600 rows for mamba stage  */
#define BQ (Bb * N_QRY)           /* 16384 rows for query stage */

__device__ __forceinline__ float siluf(float x)  { return x / (1.f + __expf(-x)); }
__device__ __forceinline__ float softplusf(float x){ return x > 20.f ? x : log1pf(__expf(x)); }
__device__ __forceinline__ float geluf(float g) {
    const float c = 0.7978845608028654f;             // sqrt(2/pi), tanh approx (jax default)
    return 0.5f * g * (1.f + tanhf(c * (g + 0.044715f * g * g * g)));
}

// ---------------------------------------------------------------------------
// Generic WMMA GEMM:  Cout = alpha * A(MxK) * B(KxN) [+ bias] [+ Cin]
// flags: 1 = add bias[n], 2 = accumulate Cin, 4 = B stored transposed [N,K]
// Two-level batching: z -> (z/H, z%H) with outer/inner strides, for (b, head).
// Block = 256 threads (8 waves), tile 128x128, K-step 32, bf16 staged in LDS.
// Wave grid 4(m) x 2(n): per wave 2 A-frags x 4 B-frags = 8 WMMAs per K-step.
// Interior tiles use unguarded b128 loads + packed bf16 converts; edge tiles
// fall back to guarded scalar staging.
// ---------------------------------------------------------------------------
__global__ __launch_bounds__(256) void k_gemm(
    const float* __restrict__ A, const float* __restrict__ Bm,
    const float* __restrict__ bias, const float* __restrict__ Cin,
    float* __restrict__ Cout,
    int M, int N, int K, int lda, int ldb, int ldc,
    int H, long long sAo, long long sAi, long long sBo, long long sBi,
    long long sCo, long long sCi, float alpha, int flags)
{
    __shared__ __align__(16) __bf16 As[128 * 34];   // [row m][k], pad to 34
    __shared__ __align__(16) __bf16 Bs[128 * 34];   // [col n][k], pad to 34

    const int z  = blockIdx.z;
    const int zo = z / H, zi = z % H;
    A  += (long long)zo * sAo + (long long)zi * sAi;
    Bm += (long long)zo * sBo + (long long)zi * sBi;
    const long long cofs = (long long)zo * sCo + (long long)zi * sCi;
    Cout += cofs;
    if (Cin) Cin += cofs;

    const int bm0 = blockIdx.y * 128, bn0 = blockIdx.x * 128;
    const int tid = threadIdx.x, lane = tid & 31, wave = tid >> 5;
    const int half = lane >> 4, l16 = lane & 15;
    const int wm0 = (wave & 3) * 32;     // wave row offset inside block tile
    const int wn0 = (wave >> 2) * 64;    // wave col offset inside block tile
    const bool bt = (flags & 4) != 0;

    const bool aligned_a = ((lda & 3) == 0);
    const bool aligned_b = ((ldb & 3) == 0);
    const bool m_full = (bm0 + 128 <= M);
    const bool n_full = (bn0 + 128 <= N);

    v8f acc[2][4] = {};

    for (int kk = 0; kk < K; kk += 32) {
        const bool k_full = (kk + 32 <= K);

        // ---- stage A tile (128 rows x 32 k) as bf16 ------------------------
        {
            const int r = tid >> 1, kc = (tid & 1) * 16;
            const int gr = bm0 + r;
            const float* ap = A + (long long)gr * lda + kk + kc;
            if (aligned_a && m_full && k_full) {
                #pragma unroll
                for (int q = 0; q < 4; q++) {
                    vf4 v = *(const vf4*)(ap + 4 * q);
                    bf16x2* dst = (bf16x2*)&As[r * 34 + kc + 4 * q];
                    bf16x2 lo = { (__bf16)v.x, (__bf16)v.y };
                    bf16x2 hi = { (__bf16)v.z, (__bf16)v.w };
                    dst[0] = lo;
                    dst[1] = hi;
                }
            } else {
                #pragma unroll
                for (int jj = 0; jj < 16; jj++) {
                    const int gk = kk + kc + jj;
                    float v = (gr < M && gk < K) ? ap[jj] : 0.f;
                    As[r * 34 + kc + jj] = (__bf16)v;
                }
            }
        }
        // ---- stage B tile as Bs[n][k] --------------------------------------
        if (!bt) {
            const int kr = tid >> 3, nc = (tid & 7) * 16;
            const int gk = kk + kr;
            const float* bp = Bm + (long long)gk * ldb + bn0 + nc;
            if (aligned_b && n_full && k_full) {
                #pragma unroll
                for (int q = 0; q < 4; q++) {
                    vf4 v = *(const vf4*)(bp + 4 * q);
                    Bs[(nc + 4 * q + 0) * 34 + kr] = (__bf16)v.x;
                    Bs[(nc + 4 * q + 1) * 34 + kr] = (__bf16)v.y;
                    Bs[(nc + 4 * q + 2) * 34 + kr] = (__bf16)v.z;
                    Bs[(nc + 4 * q + 3) * 34 + kr] = (__bf16)v.w;
                }
            } else {
                #pragma unroll
                for (int jj = 0; jj < 16; jj++) {
                    const int gn = bn0 + nc + jj;
                    float v = (gk < K && gn < N) ? bp[jj] : 0.f;
                    Bs[(nc + jj) * 34 + kr] = (__bf16)v;
                }
            }
        } else {
            const int r = tid >> 1, kc = (tid & 1) * 16;
            const int gn = bn0 + r;
            const float* bp = Bm + (long long)gn * ldb + kk + kc;
            if (aligned_b && n_full && k_full) {
                #pragma unroll
                for (int q = 0; q < 4; q++) {
                    vf4 v = *(const vf4*)(bp + 4 * q);
                    bf16x2* dst = (bf16x2*)&Bs[r * 34 + kc + 4 * q];
                    bf16x2 lo = { (__bf16)v.x, (__bf16)v.y };
                    bf16x2 hi = { (__bf16)v.z, (__bf16)v.w };
                    dst[0] = lo;
                    dst[1] = hi;
                }
            } else {
                #pragma unroll
                for (int jj = 0; jj < 16; jj++) {
                    const int gk = kk + kc + jj;
                    float v = (gn < N && gk < K) ? bp[jj] : 0.f;
                    Bs[r * 34 + kc + jj] = (__bf16)v;
                }
            }
        }
        // CDNA5 prefetch of the next K tile (global_prefetch_b8)
        if (kk + 32 < K) {
            __builtin_prefetch(A + (long long)(bm0 + (tid >> 1)) * lda + kk + 32, 0, 1);
            __builtin_prefetch(Bm + (bt ? ((long long)(bn0 + (tid >> 1)) * ldb + kk + 32)
                                        : ((long long)(kk + 32 + (tid >> 3)) * ldb + bn0)),
                               0, 1);
        }
        __syncthreads();

        // ---- A fragments: 16-bit A layout (lane m=l16, half selects K phase)
        union { v16bf v; unsigned u[8]; } af[2];
        #pragma unroll
        for (int mi = 0; mi < 2; mi++) {
            const int arow = (wm0 + mi * 16 + l16) * 34;
            #pragma unroll
            for (int v = 0; v < 8; v++) {
                const int k0 = ((v < 4) ? 0 : 16) + 8 * half + 2 * (v & 3);
                af[mi].u[v] = *(const unsigned*)&As[arow + k0];
            }
        }
        // ---- 2x4 WMMAs over the 32x64 wave tile ----------------------------
        #pragma unroll
        for (int j = 0; j < 4; j++) {
            union { v16bf v; unsigned u[8]; } bf;
            const int brow = (wn0 + j * 16 + l16) * 34;
            #pragma unroll
            for (int v = 0; v < 8; v++)
                bf.u[v] = *(const unsigned*)&Bs[brow + ((v < 4) ? 0 : 16) + 8 * half + 2 * (v & 3)];
            #pragma unroll
            for (int mi = 0; mi < 2; mi++)
                acc[mi][j] = __builtin_amdgcn_wmma_f32_16x16x32_bf16(
                    false, af[mi].v, false, bf.v, (short)0, acc[mi][j], false, false);
        }
        __syncthreads();
    }

    // ---- epilogue: C layout VGPR r -> (m = r + 8*half, n = l16) ------------
    #pragma unroll
    for (int j = 0; j < 4; j++) {
        const int col = bn0 + wn0 + j * 16 + l16;
        if (col >= N) continue;
        const float bv = (flags & 1) ? bias[col] : 0.f;
        #pragma unroll
        for (int mi = 0; mi < 2; mi++) {
            #pragma unroll
            for (int r = 0; r < 8; r++) {
                const int row = bm0 + wm0 + mi * 16 + r + 8 * half;
                if (row >= M) continue;
                float v = alpha * acc[mi][j][r] + bv;
                const long long ci = (long long)row * ldc + col;
                if (flags & 2) v += Cin[ci];
                Cout[ci] = v;
            }
        }
    }
}

// --------------------------- small VALU kernels ----------------------------

__global__ void k_rmsnorm(const float* __restrict__ x, const float* __restrict__ w,
                          float* __restrict__ o, int rows, int dim)
{
    const int row = blockIdx.x * (blockDim.x >> 5) + (threadIdx.x >> 5);
    const int lane = threadIdx.x & 31;
    if (row >= rows) return;
    const float* xr = x + (long long)row * dim;
    float s = 0.f;
    for (int i = lane; i < dim; i += 32) { float v = xr[i]; s += v * v; }
    #pragma unroll
    for (int o2 = 16; o2; o2 >>= 1) s += __shfl_xor(s, o2, 32);
    const float r = rsqrtf(s / dim + 1e-5f);
    float* orow = o + (long long)row * dim;
    for (int i = lane; i < dim; i += 32) orow[i] = xr[i] * r * w[i];
}

__global__ void k_layernorm(const float* __restrict__ x, const float* __restrict__ w,
                            const float* __restrict__ b, float* __restrict__ o,
                            int rows, int dim)
{
    const int row = blockIdx.x * (blockDim.x >> 5) + (threadIdx.x >> 5);
    const int lane = threadIdx.x & 31;
    if (row >= rows) return;
    const float* xr = x + (long long)row * dim;
    float s = 0.f, s2 = 0.f;
    for (int i = lane; i < dim; i += 32) { float v = xr[i]; s += v; s2 += v * v; }
    #pragma unroll
    for (int o2 = 16; o2; o2 >>= 1) { s += __shfl_xor(s, o2, 32); s2 += __shfl_xor(s2, o2, 32); }
    const float m = s / dim;
    const float r = rsqrtf(s2 / dim - m * m + 1e-5f);
    float* orow = o + (long long)row * dim;
    for (int i = lane; i < dim; i += 32) orow[i] = (xr[i] - m) * r * w[i] + b[i];
}

// depthwise causal conv (D_CONV=4) + silu; xz row stride 2*D_INNER, xs = cols [0,800)
__global__ void k_conv_silu(const float* __restrict__ xz, const float* __restrict__ cw,
                            const float* __restrict__ cb, float* __restrict__ xc)
{
    const int idx = blockIdx.x * blockDim.x + threadIdx.x;
    if (idx >= Bb * N_ANCHOR * D_INNER) return;
    const int d = idx % D_INNER;
    const int t = (idx / D_INNER) % N_ANCHOR;
    const int b = idx / (D_INNER * N_ANCHOR);
    float acc = cb[d];
    #pragma unroll
    for (int k = 0; k < 4; k++) {
        const int tt = t + k - 3;
        if (tt >= 0)
            acc += xz[((long long)(b * N_ANCHOR + tt)) * (2 * D_INNER) + d] * cw[d * 4 + k];
    }
    xc[idx] = siluf(acc);
}

__global__ void k_softplus(float* __restrict__ x, int n)
{
    const int i = blockIdx.x * blockDim.x + threadIdx.x;
    if (i < n) x[i] = softplusf(x[i]);
}

// selective scan: one thread per (b, d, n); 16-lane butterfly reduce over n.
// dbc has ldc=64: dt part [0,25), Bm [25,41), Cm [41,57).
__global__ void k_scan(const float* __restrict__ dt, const float* __restrict__ xc,
                       const float* __restrict__ dbc, const float* __restrict__ A_log,
                       const float* __restrict__ Dw, float* __restrict__ y)
{
    const int idx = blockIdx.x * blockDim.x + threadIdx.x;
    if (idx >= Bb * D_INNER * D_STATE) return;
    const int n = idx & 15;
    const int d = (idx >> 4) % D_INNER;
    const int b = idx / (D_STATE * D_INNER);
    const float a = -__expf(A_log[d * D_STATE + n]);
    float h = 0.f;
    const int base = b * N_ANCHOR;
    for (int t = 0; t < N_ANCHOR; t++) {
        const long long r = base + t;
        const float dtv = dt[r * D_INNER + d];
        const float xcv = xc[r * D_INNER + d];
        const float bm  = dbc[r * 64 + DT_RANK + n];
        const float cm  = dbc[r * 64 + DT_RANK + D_STATE + n];
        h = __expf(dtv * a) * h + dtv * xcv * bm;
        float p = h * cm;
        #pragma unroll
        for (int o = 8; o; o >>= 1) p += __shfl_xor(p, o, 16);
        if (n == 0) y[r * D_INNER + d] = p + xcv * Dw[d];
    }
}

__global__ void k_gate(float* __restrict__ y, const float* __restrict__ xz, int n)
{
    const int i = blockIdx.x * blockDim.x + threadIdx.x;
    if (i >= n) return;
    const int r = i / D_INNER, d = i % D_INNER;
    y[i] *= siluf(xz[(long long)r * (2 * D_INNER) + D_INNER + d]);
}

// positional features: pe[.,0:24)=sin, [24,48)=cos, [48,51)=qry, [51,64)=0
__global__ void k_pe(const float* __restrict__ qry, float* __restrict__ pe)
{
    const int i = blockIdx.x * blockDim.x + threadIdx.x;
    if (i >= BQ) return;
    const float* q = qry + (long long)i * 3;
    float* p = pe + (long long)i * 64;
    #pragma unroll
    for (int e = 0; e < 24; e++) {
        const float f = q[e >> 3] * (3.14159265358979323846f * (float)(1 << (e & 7)));
        p[e]      = __sinf(f);
        p[24 + e] = __cosf(f);
    }
    p[48] = q[0]; p[49] = q[1]; p[50] = q[2];
    #pragma unroll
    for (int c = 51; c < 64; c++) p[c] = 0.f;
}

__global__ void k_pad_pew(const float* __restrict__ pw, float* __restrict__ pwp)
{
    const int i = blockIdx.x * blockDim.x + threadIdx.x;
    if (i >= 64 * D_MODEL) return;
    const int r = i / D_MODEL, c = i % D_MODEL;
    pwp[i] = (r < 51) ? pw[r * D_MODEL + c] : 0.f;
}

__global__ void k_softmax(float* __restrict__ s, int rows, int dim)
{
    const int row = blockIdx.x * (blockDim.x >> 5) + (threadIdx.x >> 5);
    const int lane = threadIdx.x & 31;
    if (row >= rows) return;
    float* sr = s + (long long)row * dim;
    float m = -3.4e38f;
    for (int i = lane; i < dim; i += 32) m = fmaxf(m, sr[i]);
    #pragma unroll
    for (int o2 = 16; o2; o2 >>= 1) m = fmaxf(m, __shfl_xor(m, o2, 32));
    float sum = 0.f;
    for (int i = lane; i < dim; i += 32) { const float e = __expf(sr[i] - m); sr[i] = e; sum += e; }
    #pragma unroll
    for (int o2 = 16; o2; o2 >>= 1) sum += __shfl_xor(sum, o2, 32);
    const float inv = 1.f / sum;
    for (int i = lane; i < dim; i += 32) sr[i] *= inv;
}

__global__ void k_geglu(const float* __restrict__ h, float* __restrict__ o, int rows, int half)
{
    const int i = blockIdx.x * blockDim.x + threadIdx.x;
    if (i >= rows * half) return;
    const int r = i / half, j = i % half;
    const float a = h[(long long)r * (2 * half) + j];
    const float g = h[(long long)r * (2 * half) + half + j];
    o[i] = a * geluf(g);
}

__global__ void k_final(const float* __restrict__ lat, const float* __restrict__ ow,
                        const float* __restrict__ ob, float* __restrict__ out,
                        int rows, int dim)
{
    const int row = blockIdx.x * (blockDim.x >> 5) + (threadIdx.x >> 5);
    const int lane = threadIdx.x & 31;
    if (row >= rows) return;
    const float* lr = lat + (long long)row * dim;
    float s = 0.f;
    for (int i = lane; i < dim; i += 32) s += lr[i] * ow[i];
    #pragma unroll
    for (int o2 = 16; o2; o2 >>= 1) s += __shfl_xor(s, o2, 32);
    if (lane == 0) out[row] = s + ob[0];
}

// ------------------------------ host driver --------------------------------

static void gemm(hipStream_t s, const float* A, const float* B, const float* bias,
                 const float* Cin, float* Cout, int M, int N, int K,
                 int lda, int ldb, int ldc, float alpha, int flags,
                 int batches = 1, int H = 1,
                 long long sAo = 0, long long sAi = 0,
                 long long sBo = 0, long long sBi = 0,
                 long long sCo = 0, long long sCi = 0)
{
    dim3 g((N + 127) / 128, (M + 127) / 128, batches), b(256);
    k_gemm<<<g, b, 0, s>>>(A, B, bias, Cin, Cout, M, N, K, lda, ldb, ldc,
                           H, sAo, sAi, sBo, sBi, sCo, sCi, alpha, flags);
}

extern "C" void kernel_launch(void* const* d_in, const int* in_sizes, int n_in,
                              void* d_out, int out_size, void* d_ws, size_t ws_size,
                              hipStream_t stream)
{
    (void)in_sizes; (void)n_in; (void)out_size; (void)ws_size;
    auto F = [&](int i) { return (const float*)d_in[i]; };

    // jax pytree flatten order (dict keys sorted, list order preserved):
    const float* mash    = F(0);
    const float* qry     = F(1);
    const float* lnc_b   = F(2);
    const float* lnc_w   = F(3);
    const float* lnq_b   = F(4);
    const float* lnq_w   = F(5);
    const float* ca_ob   = F(6);
    const float* ca_ow   = F(7);
    const float* to_kv   = F(8);
    const float* to_q    = F(9);
    const float* ff_b1   = F(10);
    const float* ff_b2   = F(11);
    const float* ff_lnb  = F(12);
    const float* ff_lnw  = F(13);
    const float* ff_w1   = F(14);
    const float* ff_w2   = F(15);
    const int LBASE = 16;                       // 10 leaves per layer x 24
    const float* out_b   = F(LBASE + 10 * N_LAYER + 0);
    const float* out_w   = F(LBASE + 10 * N_LAYER + 1);
    const float* pe_b    = F(LBASE + 10 * N_LAYER + 2);
    const float* pe_w    = F(LBASE + 10 * N_LAYER + 3);
    const float* proj_b  = F(LBASE + 10 * N_LAYER + 4);
    const float* proj_w  = F(LBASE + 10 * N_LAYER + 5);

    // workspace layout (floats), with lifetime-based reuse
    float* ws  = (float*)d_ws;
    float* x    = ws;                          // 1600x400
    float* xn   = x    + 640000;               // 1600x400 (rms/ln scratch)
    float* xz   = xn   + 640000;               // 1600x1600
    float* xc   = xz   + 2560000;              // 1600x800
    float* dbc  = xc   + 1280000;              // 1600x64 (N=57 padded)
    float* dtb  = dbc  + 102400;               // 1600x800
    float* yb   = dtb  + 1280000;              // 1600x800
    float* kv   = xz;                          // reuse mamba scratch: 1600x3200
    float* pe   = yb   + 1280000;              // 16384x64
    float* pwp  = pe   + 1048576;              // 64x400
    float* qe   = pwp  + 25600;                // 16384x400
    float* qn   = qe   + 6553600;              // 16384x400
    float* qb   = qn   + 6553600;              // 16384x1600
    float* sim  = qb   + 26214400;             // 16x4096x400
    float* att  = sim  + 26214400;             // 16384x1600
    float* lat  = att  + 26214400;             // 16384x400
    float* latn = lat  + 6553600;              // 16384x400
    float* ffh  = qb;                          // reuse qb+sim: 16384x3200
    float* ffa  = att;                         // reuse att:   16384x1600

    // ---- anchor embedding: x = mash @ proj_w + proj_b  (K=22 via guards) ---
    gemm(stream, mash, proj_w, proj_b, nullptr, x, BL, D_MODEL, 22, 22, D_MODEL, D_MODEL, 1.f, 1);

    // ---- 24 Mamba layers ---------------------------------------------------
    for (int l = 0; l < N_LAYER; l++) {
        const float* A_log  = F(LBASE + 10 * l + 0);
        const float* Dw     = F(LBASE + 10 * l + 1);
        const float* conv_b = F(LBASE + 10 * l + 2);
        const float* conv_w = F(LBASE + 10 * l + 3);
        const float* dt_b   = F(LBASE + 10 * l + 4);
        const float* dt_w   = F(LBASE + 10 * l + 5);
        const float* in_pj  = F(LBASE + 10 * l + 6);
        const float* norm_w = F(LBASE + 10 * l + 7);
        const float* out_pj = F(LBASE + 10 * l + 8);
        const float* x_pj   = F(LBASE + 10 * l + 9);

        k_rmsnorm<<<BL / 8, 256, 0, stream>>>(x, norm_w, xn, BL, D_MODEL);
        gemm(stream, xn, in_pj, nullptr, nullptr, xz, BL, 2 * D_INNER, D_MODEL,
             D_MODEL, 2 * D_INNER, 2 * D_INNER, 1.f, 0);
        k_conv_silu<<<(BL * D_INNER + 255) / 256, 256, 0, stream>>>(xz, conv_w, conv_b, xc);
        gemm(stream, xc, x_pj, nullptr, nullptr, dbc, BL, DT_RANK + 2 * D_STATE, D_INNER,
             D_INNER, DT_RANK + 2 * D_STATE, 64, 1.f, 0);
        gemm(stream, dbc, dt_w, dt_b, nullptr, dtb, BL, D_INNER, DT_RANK,
             64, D_INNER, D_INNER, 1.f, 1);
        k_softplus<<<(BL * D_INNER + 255) / 256, 256, 0, stream>>>(dtb, BL * D_INNER);
        k_scan<<<(Bb * D_INNER * D_STATE) / 256, 256, 0, stream>>>(dtb, xc, dbc, A_log, Dw, yb);
        k_gate<<<(BL * D_INNER + 255) / 256, 256, 0, stream>>>(yb, xz, BL * D_INNER);
        gemm(stream, yb, out_pj, nullptr, x, x, BL, D_MODEL, D_INNER,
             D_INNER, D_MODEL, D_MODEL, 1.f, 2 /*accum residual*/);
    }

    // ---- query positional embedding (K padded 51 -> 64, stays on WMMA) -----
    k_pe<<<(BQ + 255) / 256, 256, 0, stream>>>(qry, pe);
    k_pad_pew<<<(64 * D_MODEL + 255) / 256, 256, 0, stream>>>(pe_w, pwp);
    gemm(stream, pe, pwp, pe_b, nullptr, qe, BQ, D_MODEL, 64, 64, D_MODEL, D_MODEL, 1.f, 1);

    // ---- cross attention ---------------------------------------------------
    k_layernorm<<<BQ / 8, 256, 0, stream>>>(qe, lnq_w, lnq_b, qn, BQ, D_MODEL);
    gemm(stream, qn, to_q, nullptr, nullptr, qb, BQ, HEADS * DIM_HEAD, D_MODEL,
         D_MODEL, HEADS * DIM_HEAD, HEADS * DIM_HEAD, 1.f, 0);
    k_layernorm<<<BL / 8, 256, 0, stream>>>(x, lnc_w, lnc_b, xn, BL, D_MODEL);
    gemm(stream, xn, to_kv, nullptr, nullptr, kv, BL, 2 * HEADS * DIM_HEAD, D_MODEL,
         D_MODEL, 2 * HEADS * DIM_HEAD, 2 * HEADS * DIM_HEAD, 1.f, 0);

    // sim[b,h,q,a] = (q . k) / sqrt(DIM_HEAD), batched over 16 (b,h), B-transposed
    gemm(stream, qb, kv, nullptr, nullptr, sim, N_QRY, N_ANCHOR, DIM_HEAD,
         HEADS * DIM_HEAD, 2 * HEADS * DIM_HEAD, N_ANCHOR, 0.05f, 4 /*BT*/,
         Bb * HEADS, HEADS,
         (long long)N_QRY * HEADS * DIM_HEAD, DIM_HEAD,
         (long long)N_ANCHOR * 2 * HEADS * DIM_HEAD, DIM_HEAD,
         (long long)HEADS * N_QRY * N_ANCHOR, (long long)N_QRY * N_ANCHOR);
    k_softmax<<<(Bb * HEADS * N_QRY) / 8, 256, 0, stream>>>(sim, Bb * HEADS * N_QRY, N_ANCHOR);
    // att[b,q,h*400+d] = sum_a sim * v
    gemm(stream, sim, kv + HEADS * DIM_HEAD, nullptr, nullptr, att,
         N_QRY, DIM_HEAD, N_ANCHOR, N_ANCHOR, 2 * HEADS * DIM_HEAD, HEADS * DIM_HEAD, 1.f, 0,
         Bb * HEADS, HEADS,
         (long long)HEADS * N_QRY * N_ANCHOR, (long long)N_QRY * N_ANCHOR,
         (long long)N_ANCHOR * 2 * HEADS * DIM_HEAD, DIM_HEAD,
         (long long)N_QRY * HEADS * DIM_HEAD, DIM_HEAD);
    gemm(stream, att, ca_ow, ca_ob, nullptr, lat, BQ, D_MODEL, HEADS * DIM_HEAD,
         HEADS * DIM_HEAD, D_MODEL, D_MODEL, 1.f, 1);

    // ---- GEGLU feed-forward with residual ---------------------------------
    k_layernorm<<<BQ / 8, 256, 0, stream>>>(lat, ff_lnw, ff_lnb, latn, BQ, D_MODEL);
    gemm(stream, latn, ff_w1, ff_b1, nullptr, ffh, BQ, 2 * 4 * D_MODEL, D_MODEL,
         D_MODEL, 2 * 4 * D_MODEL, 2 * 4 * D_MODEL, 1.f, 1);
    k_geglu<<<((long long)BQ * 4 * D_MODEL + 255) / 256, 256, 0, stream>>>(ffh, ffa, BQ, 4 * D_MODEL);
    gemm(stream, ffa, ff_w2, ff_b2, lat, lat, BQ, D_MODEL, 4 * D_MODEL,
         4 * D_MODEL, D_MODEL, D_MODEL, 1.f, 3 /*bias|accum*/);

    // ---- scalar head -------------------------------------------------------
    k_final<<<BQ / 8, 256, 0, stream>>>(lat, out_w, out_b, (float*)d_out, BQ, D_MODEL);
}
